// DeepseekV2Attention_4664334484016
// MI455X (gfx1250) — matrix operand order repaired and checked by reading, not compiled, not run on hardware
//
#include <hip/hip_runtime.h>

// ---------------------------------------------------------------------------
// DeepSeek-V2 MLA attention for MI455X (gfx1250, wave32, WMMA, TDM).
//   q   = x @ w_q                        (WMMA GEMM, f32 out)
//   kv  = x @ w_kv_a                     (WMMA GEMM, f32 out)
//   kv_post: rmsnorm latent + rope k_pe -> k_full bf16 (S,576), latentT bf16 (512,S)
//   q_split: split q, rope q_pe          -> qn bf16 (H,S,128), q_full[...,512:576]
//   q_abs = qn_h @ kc_h                  (batched WMMA GEMM, bf16 out)
//   flash attention: K blocks staged to LDS by the Tensor Data Mover
//     (tensor_load_to_lds, double buffered, s_wait_tensorcnt + barrier),
//     online softmax, P via LDS relayout, PV against latentT.
//   attn_cat = o_h @ vc_h, out = attn_cat @ w_o   (WMMA GEMMs)
// ---------------------------------------------------------------------------

typedef __attribute__((ext_vector_type(16))) __bf16 v16bf;
typedef __attribute__((ext_vector_type(8)))  __bf16 v8bf;
typedef __attribute__((ext_vector_type(8)))  float  v8f;
typedef __attribute__((ext_vector_type(4)))  unsigned int v4u;
typedef __attribute__((ext_vector_type(8)))  int v8i;
typedef __attribute__((ext_vector_type(4)))  int v4i;

#define DEV __device__ __forceinline__

constexpr int   kS    = 2048;
constexpr int   kHID  = 2048;
constexpr int   kH    = 16;
constexpr int   kNOPE = 128;
constexpr int   kROPE = 64;
constexpr int   kKVR  = 512;
constexpr int   kVH   = 128;
constexpr int   kD    = kKVR + kROPE;          // 576
constexpr int   kQD   = kNOPE + kROPE;         // 192
constexpr float kSCALE = 0.07216878364870323f; // 192^-0.5
constexpr float kEPS   = 1e-6f;

// ---------------- WMMA fragment loaders ------------------------------------
DEV v16bf load_a_frag(const __bf16* __restrict__ A, int lda, int row, int k0, int lane) {
    int kb = k0 + ((lane >> 4) << 3);
    const __bf16* p = A + (size_t)row * lda + kb;
    v8bf lo = *(const v8bf*)(p);
    v8bf hi = *(const v8bf*)(p + 16);
    v16bf f;
#pragma unroll
    for (int i = 0; i < 8; ++i) { f[i] = lo[i]; f[i + 8] = hi[i]; }
    return f;
}

DEV v16bf load_b_frag(const __bf16* __restrict__ Bt, int ldb, int col, int k0, int lane) {
    int kb = k0 + ((lane >> 4) << 3);
    const __bf16* p = Bt + (size_t)col * ldb + kb;
    v8bf lo = *(const v8bf*)(p);
    v8bf hi = *(const v8bf*)(p + 16);
    v16bf f;
#pragma unroll
    for (int i = 0; i < 8; ++i) { f[i] = lo[i]; f[i + 8] = hi[i]; }
    return f;
}

DEV v8f wmma_bf16(v16bf a, v16bf b, v8f c) {
    return __builtin_amdgcn_wmma_f32_16x16x32_bf16(false, a, false, b, (short)0, c,
                                                   false, false);
}

DEV float rowmax16(float v) {
#pragma unroll
    for (int off = 8; off; off >>= 1) v = fmaxf(v, __shfl_xor(v, off, 16));
    return v;
}
DEV float rowsum16(float v) {
#pragma unroll
    for (int off = 8; off; off >>= 1) v += __shfl_xor(v, off, 16);
    return v;
}

// ---------------- TDM: 2-D tile global -> LDS (D# per CDNA5 ISA ch.8) -------
// data_size = 2 bytes; dims/strides in elements. Groups 2/3 zero (<=2D tile).
// This toolchain exposes the 6-arg builtin: (v4u g0, v8i g1, v4i g2, v4i g3,
// v8i extra, i32 cpol).
DEV void tdm_load_2d(unsigned lds_addr, const void* gaddr,
                     unsigned tile_d0, unsigned tile_d1,
                     unsigned tensor_d0, unsigned tensor_d1,
                     unsigned long stride0) {
    unsigned long ga = (unsigned long)gaddr;
    v4u g0;
    g0[0] = 1u;                                             // count=1, user mode
    g0[1] = lds_addr;                                       // lds byte address
    g0[2] = (unsigned)(ga & 0xffffffffu);                   // global_addr[31:0]
    g0[3] = (unsigned)((ga >> 32) & 0x01ffffffu) | (2u << 30); // addr[56:32]|type=2
    v8i g1;
    g1[0] = (int)(1u << 16);                                // data_size=1 (2 bytes)
    g1[1] = (int)((tensor_d0 & 0xffffu) << 16);             // tensor_dim0 lo
    g1[2] = (int)((tensor_d0 >> 16) | ((tensor_d1 & 0xffffu) << 16));
    g1[3] = (int)((tensor_d1 >> 16) | ((tile_d0 & 0xffffu) << 16));
    g1[4] = (int)(tile_d1 & 0xffffu);                       // tile_dim1 (tile_dim2=0)
    g1[5] = (int)(stride0 & 0xffffffffu);                   // dim0 stride lo
    g1[6] = (int)((stride0 >> 32) & 0xffffu);               // stride hi (dim1 stride=0)
    g1[7] = 0;
    v4i gz4 = {0, 0, 0, 0};
    v8i gz8 = {0, 0, 0, 0, 0, 0, 0, 0};
    __builtin_amdgcn_tensor_load_to_lds(g0, g1, gz4, gz4, gz8, 0);
}

// ---------------- elementwise helpers ---------------------------------------
__global__ void cast_bf16_kernel(const float* __restrict__ src, __bf16* __restrict__ dst,
                                 int n) {
    int i = blockIdx.x * 256 + threadIdx.x;
    if (i < n) dst[i] = (__bf16)src[i];
}

__global__ void castT_kernel(const float* __restrict__ src, __bf16* __restrict__ dst,
                             int R, int C) {
    size_t base = (size_t)blockIdx.z * R * C;
    int i = blockIdx.x * 256 + threadIdx.x;
    if (i < R * C) {
        int r = i / C, c = i % C;
        dst[base + (size_t)c * R + r] = (__bf16)src[base + i];
    }
}

// ---------------- generic NT WMMA GEMM --------------------------------------
template <typename OutT>
__global__ __launch_bounds__(256) void gemm_nt_kernel(
    const __bf16* __restrict__ A, const __bf16* __restrict__ Bt, OutT* __restrict__ C,
    int M, int N, int K, int lda, int ldb, int ldc,
    long strideA, long strideB, long strideC) {
    int b = blockIdx.z;
    A  += (size_t)b * strideA;
    Bt += (size_t)b * strideB;
    C  += (size_t)b * strideC;

    int lane = threadIdx.x & 31;
    int w    = threadIdx.x >> 5;
    int m0   = blockIdx.y * 256 + w * 32;
    int n0   = blockIdx.x * 64;
    if (m0 >= M) return;

    v8f acc[2][4];
#pragma unroll
    for (int s = 0; s < 2; ++s)
#pragma unroll
        for (int t = 0; t < 4; ++t)
#pragma unroll
            for (int r = 0; r < 8; ++r) acc[s][t][r] = 0.0f;

    int arow = m0 + (lane & 15);
    int ncol = lane & 15;
    for (int k0 = 0; k0 < K; k0 += 32) {
        v16bf af0 = load_a_frag(A, lda, arow, k0, lane);
        v16bf af1 = load_a_frag(A, lda, arow + 16, k0, lane);
#pragma unroll
        for (int t = 0; t < 4; ++t) {
            v16bf bf_ = load_b_frag(Bt, ldb, n0 + t * 16 + ncol, k0, lane);
            acc[0][t] = wmma_bf16(af0, bf_, acc[0][t]);
            acc[1][t] = wmma_bf16(af1, bf_, acc[1][t]);
        }
    }

#pragma unroll
    for (int s = 0; s < 2; ++s) {
        int mbase = m0 + s * 16 + 8 * (lane >> 4);
#pragma unroll
        for (int t = 0; t < 4; ++t)
#pragma unroll
            for (int r = 0; r < 8; ++r)
                C[(size_t)(mbase + r) * ldc + n0 + t * 16 + ncol] = (OutT)acc[s][t][r];
    }
}

// ---------------- kv epilogue: rmsnorm + rope -------------------------------
__global__ void kv_post_kernel(const float* __restrict__ kvf,
                               const float* __restrict__ norm_w,
                               const float* __restrict__ cosT,
                               const float* __restrict__ sinT,
                               __bf16* __restrict__ k_full,
                               __bf16* __restrict__ latentT) {
    int s = blockIdx.x * 8 + (threadIdx.x >> 5);
    int lane = threadIdx.x & 31;
    const float* row = kvf + (size_t)s * kD;

    float ss = 0.f;
#pragma unroll
    for (int i = 0; i < 16; ++i) { float v = row[lane * 16 + i]; ss += v * v; }
#pragma unroll
    for (int off = 16; off; off >>= 1) ss += __shfl_xor(ss, off, 32);
    float rms = rsqrtf(ss / (float)kKVR + kEPS);

#pragma unroll
    for (int i = 0; i < 16; ++i) {
        int d = lane * 16 + i;
        float v = row[d] * rms * norm_w[d];
        k_full[(size_t)s * kD + d]  = (__bf16)v;
        latentT[(size_t)d * kS + s] = (__bf16)v;
    }
    float x0 = row[kKVR + lane], x1 = row[kKVR + lane + 32];
    float c0 = cosT[s * kROPE + lane],      s0 = sinT[s * kROPE + lane];
    float c1 = cosT[s * kROPE + lane + 32], s1 = sinT[s * kROPE + lane + 32];
    k_full[(size_t)s * kD + kKVR + lane]      = (__bf16)(x0 * c0 - x1 * s0);
    k_full[(size_t)s * kD + kKVR + lane + 32] = (__bf16)(x1 * c1 + x0 * s1);
}

// ---------------- q split + rope --------------------------------------------
__global__ void q_split_kernel(const float* __restrict__ qf,
                               const float* __restrict__ cosT,
                               const float* __restrict__ sinT,
                               __bf16* __restrict__ qn,
                               __bf16* __restrict__ q_full) {
    int s = blockIdx.x, h = blockIdx.y, t = threadIdx.x; // block = 128
    const float* src = qf + (size_t)s * (kH * kQD) + h * kQD;
    qn[((size_t)h * kS + s) * kNOPE + t] = (__bf16)src[t];
    if (t < kROPE) {
        float x  = src[kNOPE + t];
        float xo = (t < 32) ? src[kNOPE + t + 32] : src[kNOPE + t - 32];
        float c  = cosT[s * kROPE + t], sn = sinT[s * kROPE + t];
        float v  = (t < 32) ? (x * c - xo * sn) : (x * c + xo * sn);
        q_full[((size_t)h * kS + s) * kD + kKVR + t] = (__bf16)v;
    }
}

// ---------------- flash attention with TDM-staged K -------------------------
// grid (S/128, H), block 256 (8 waves); wave w owns query rows [qm, qm+16).
// Wave 0 DMAs each 32x576 K block into LDS with tensor_load_to_lds (double
// buffered); all waves consume B-frags from LDS. PV runs against latentT in
// global (prefetched one block ahead).
__global__ __launch_bounds__(256) void mla_flash_kernel(
    const __bf16* __restrict__ q_full,   // (H,S,576)
    const __bf16* __restrict__ k_full,   // (S,576)
    const __bf16* __restrict__ latentT,  // (512,S)
    __bf16* __restrict__ o_bf) {         // (H,S,512)
    int h    = blockIdx.y;
    int lane = threadIdx.x & 31;
    int w    = threadIdx.x >> 5;
    int blockBase = blockIdx.x * 128;
    int qm   = blockBase + w * 16;
    int hl   = lane >> 4;
    int ncol = lane & 15;

    const __bf16* Q = q_full + (size_t)h * kS * kD;

    __shared__ __bf16 kbuf[2][32 * kD];     // 2 x 36 KB
    __shared__ __bf16 pbuf[8][16 * 32];     // 8 KB
    __bf16* pw = pbuf[w];

    v8f acc[32];
#pragma unroll
    for (int t = 0; t < 32; ++t)
#pragma unroll
        for (int r = 0; r < 8; ++r) acc[t][r] = 0.0f;

    float mrow[8], lrow[8];
#pragma unroll
    for (int r = 0; r < 8; ++r) { mrow[r] = -1e30f; lrow[r] = 0.0f; }

    int arow = qm + ncol;
    int jmax = blockBase + 128;

    // preload K block 0 via the tensor data mover
    if (w == 0) {
        tdm_load_2d((unsigned)(size_t)&kbuf[0][0], k_full, kD, 32, kD, 32,
                    (unsigned long)kD);
        __builtin_amdgcn_s_wait_tensorcnt(0);
    }
    __syncthreads();

    for (int jt = 0; jt < jmax; jt += 32) {
        int buf = (jt >> 5) & 1;
        // kick off DMA of the next K block into the other buffer
        if (w == 0 && jt + 32 < jmax)
            tdm_load_2d((unsigned)(size_t)&kbuf[buf ^ 1][0],
                        k_full + (size_t)(jt + 32) * kD, kD, 32, kD, 32,
                        (unsigned long)kD);
        // prefetch next latentT column block (each thread covers 2 rows)
        if (jt + 32 < jmax) {
            __builtin_prefetch(
                (const void*)(latentT + (size_t)(threadIdx.x * 2) * kS + jt + 32), 0, 1);
            __builtin_prefetch(
                (const void*)(latentT + (size_t)(threadIdx.x * 2 + 1) * kS + jt + 32), 0,
                1);
        }

        if (jt < qm + 16) {  // causal: this wave still needs this key block
            const __bf16* kb = &kbuf[buf][0];

            // ---- scores: 2 tiles of 16x16, K=576 (A from global, B from LDS)
            v8f sc[2];
#pragma unroll
            for (int c = 0; c < 2; ++c) {
                v8f s;
#pragma unroll
                for (int r = 0; r < 8; ++r) s[r] = 0.0f;
#pragma unroll
                for (int kc = 0; kc < 18; ++kc) {
                    v16bf af  = load_a_frag(Q, kD, arow, kc * 32, lane);
                    v16bf bf_ = load_b_frag(kb, kD, c * 16 + ncol, kc * 32, lane);
                    s = wmma_bf16(af, bf_, s);
                }
                sc[c] = s;
            }

            // ---- scale (+ mask only on the diagonal block)
            bool diag = (jt + 32 > qm);
#pragma unroll
            for (int c = 0; c < 2; ++c)
#pragma unroll
                for (int r = 0; r < 8; ++r) sc[c][r] *= kSCALE;
            if (diag) {
#pragma unroll
                for (int r = 0; r < 8; ++r) {
                    int Mg = qm + 8 * hl + r;
#pragma unroll
                    for (int c = 0; c < 2; ++c) {
                        int Ng = jt + c * 16 + ncol;
                        if (Ng > Mg) sc[c][r] = -1e30f;
                    }
                }
            }

            // ---- online softmax stats
            float alpha[8];
#pragma unroll
            for (int r = 0; r < 8; ++r) {
                float mx = rowmax16(fmaxf(sc[0][r], sc[1][r]));
                float mn = fmaxf(mrow[r], mx);
                alpha[r] = __expf(mrow[r] - mn);
                mrow[r]  = mn;
                float p0 = __expf(sc[0][r] - mn);
                float p1 = __expf(sc[1][r] - mn);
                sc[0][r] = p0;
                sc[1][r] = p1;
                lrow[r]  = lrow[r] * alpha[r] + rowsum16(p0 + p1);
            }

            // ---- rescale accumulator
#pragma unroll
            for (int t = 0; t < 32; ++t)
#pragma unroll
                for (int r = 0; r < 8; ++r) acc[t][r] *= alpha[r];

            // ---- P (C-layout f32) -> LDS -> A-layout bf16 fragment
#pragma unroll
            for (int c = 0; c < 2; ++c)
#pragma unroll
                for (int r = 0; r < 8; ++r)
                    pw[(r + 8 * hl) * 32 + c * 16 + ncol] = (__bf16)sc[c][r];
            asm volatile("s_wait_dscnt 0x0" ::: "memory"); // wave-local LDS RAW

            v16bf pa;
            {
                const __bf16* p = pw + ncol * 32 + (hl << 3);
                v8bf lo = *(const v8bf*)(p);
                v8bf hi = *(const v8bf*)(p + 16);
#pragma unroll
                for (int i = 0; i < 8; ++i) { pa[i] = lo[i]; pa[i + 8] = hi[i]; }
            }

            // ---- acc += P @ latent[jt:jt+32, :]
#pragma unroll
            for (int t = 0; t < 32; ++t) {
                v16bf bf_ = load_b_frag(latentT, kS, t * 16 + ncol, jt, lane);
                acc[t] = wmma_bf16(pa, bf_, acc[t]);
            }
        }

        if (w == 0) __builtin_amdgcn_s_wait_tensorcnt(0);
        __syncthreads();
    }

    // ---- epilogue: divide by l, write bf16
    float inv[8];
#pragma unroll
    for (int r = 0; r < 8; ++r) inv[r] = 1.0f / lrow[r];
    __bf16* O = o_bf + (size_t)h * kS * kKVR;
#pragma unroll
    for (int t = 0; t < 32; ++t)
#pragma unroll
        for (int r = 0; r < 8; ++r)
            O[(size_t)(qm + 8 * hl + r) * kKVR + t * 16 + ncol] =
                (__bf16)(acc[t][r] * inv[r]);
}

// ---------------------------------------------------------------------------
extern "C" void kernel_launch(void* const* d_in, const int* in_sizes, int n_in,
                              void* d_out, int out_size, void* d_ws, size_t ws_size,
                              hipStream_t stream) {
    const float* x       = (const float*)d_in[0]; // (1,S,HID)
    const float* cosT    = (const float*)d_in[1]; // (S,64)
    const float* sinT    = (const float*)d_in[2]; // (S,64)
    const float* w_q     = (const float*)d_in[3]; // (HID, H*192)
    const float* w_kv_a  = (const float*)d_in[4]; // (HID, 576)
    const float* norm_w  = (const float*)d_in[5]; // (512)
    const float* kc      = (const float*)d_in[6]; // (H,128,512)
    const float* vc      = (const float*)d_in[7]; // (H,512,128)
    const float* w_o     = (const float*)d_in[8]; // (H*128, HID)
    float*       out     = (float*)d_out;         // (1,S,HID)

    char* p = (char*)d_ws;
    auto alloc = [&](size_t bytes) -> void* {
        void* r = p;
        p += (bytes + 255) & ~(size_t)255;
        return r;
    };
    __bf16* xbf    = (__bf16*)alloc((size_t)kS * kHID * 2);
    __bf16* wqT    = (__bf16*)alloc((size_t)(kH * kQD) * kHID * 2);
    __bf16* wkvT   = (__bf16*)alloc((size_t)kD * kHID * 2);
    __bf16* woT    = (__bf16*)alloc((size_t)kHID * (kH * kVH) * 2);
    __bf16* kcT    = (__bf16*)alloc((size_t)kH * kKVR * kNOPE * 2);
    __bf16* vcT    = (__bf16*)alloc((size_t)kH * kVH * kKVR * 2);
    float*  qf32   = (float*)alloc((size_t)kS * (kH * kQD) * 4);
    float*  kvf32  = (float*)alloc((size_t)kS * kD * 4);
    __bf16* kfull  = (__bf16*)alloc((size_t)kS * kD * 2);
    __bf16* latT   = (__bf16*)alloc((size_t)kKVR * kS * 2);
    __bf16* qn     = (__bf16*)alloc((size_t)kH * kS * kNOPE * 2);
    __bf16* qfull  = (__bf16*)alloc((size_t)kH * kS * kD * 2);
    __bf16* obf    = (__bf16*)alloc((size_t)kH * kS * kKVR * 2);
    __bf16* acat   = (__bf16*)alloc((size_t)kS * (kH * kVH) * 2);

    {
        int n = kS * kHID;
        cast_bf16_kernel<<<(n + 255) / 256, 256, 0, stream>>>(x, xbf, n);
    }
    {
        int n = kHID * (kH * kQD);
        castT_kernel<<<dim3((n + 255) / 256, 1, 1), 256, 0, stream>>>(w_q, wqT, kHID,
                                                                      kH * kQD);
    }
    {
        int n = kHID * kD;
        castT_kernel<<<dim3((n + 255) / 256, 1, 1), 256, 0, stream>>>(w_kv_a, wkvT, kHID,
                                                                      kD);
    }
    {
        int n = (kH * kVH) * kHID;
        castT_kernel<<<dim3((n + 255) / 256, 1, 1), 256, 0, stream>>>(w_o, woT,
                                                                      kH * kVH, kHID);
    }
    {
        int n = kNOPE * kKVR;
        castT_kernel<<<dim3((n + 255) / 256, 1, kH), 256, 0, stream>>>(kc, kcT, kNOPE,
                                                                      kKVR);
    }
    {
        int n = kKVR * kVH;
        castT_kernel<<<dim3((n + 255) / 256, 1, kH), 256, 0, stream>>>(vc, vcT, kKVR,
                                                                      kVH);
    }

    gemm_nt_kernel<float><<<dim3((kH * kQD) / 64, kS / 256, 1), 256, 0, stream>>>(
        xbf, wqT, qf32, kS, kH * kQD, kHID, kHID, kHID, kH * kQD, 0, 0, 0);

    gemm_nt_kernel<float><<<dim3(kD / 64, kS / 256, 1), 256, 0, stream>>>(
        xbf, wkvT, kvf32, kS, kD, kHID, kHID, kHID, kD, 0, 0, 0);

    kv_post_kernel<<<kS / 8, 256, 0, stream>>>(kvf32, norm_w, cosT, sinT, kfull, latT);

    q_split_kernel<<<dim3(kS, kH, 1), 128, 0, stream>>>(qf32, cosT, sinT, qn, qfull);

    gemm_nt_kernel<__bf16><<<dim3(kKVR / 64, kS / 256, kH), 256, 0, stream>>>(
        qn, kcT, qfull, kS, kKVR, kNOPE, kNOPE, kNOPE, kD,
        (long)kS * kNOPE, (long)kKVR * kNOPE, (long)kS * kD);

    mla_flash_kernel<<<dim3(kS / 128, kH, 1), 256, 0, stream>>>(qfull, kfull, latT, obf);

    gemm_nt_kernel<__bf16><<<dim3(kVH / 64, kS / 256, kH), 256, 0, stream>>>(
        obf, vcT, acat, kS, kVH, kKVR, kKVR, kKVR, kH * kVH,
        (long)kS * kKVR, (long)kVH * kKVR, (long)kVH);

    gemm_nt_kernel<float><<<dim3(kHID / 64, kS / 256, 1), 256, 0, stream>>>(
        acat, woT, out, kS, kHID, kH * kVH, kH * kVH, kH * kVH, kHID, 0, 0, 0);
}
